// PixelsCoordinatesAttention_87540023428021
// MI455X (gfx1250) — matrix operand order repaired and checked by reading, not compiled, not run on hardware
//
#include <hip/hip_runtime.h>

// ---------------------------------------------------------------------------
// Types for CDNA5 WMMA (wave32): v_wmma_f32_16x16x32_bf16
// ---------------------------------------------------------------------------
typedef __attribute__((ext_vector_type(16))) __bf16 v16bf;
typedef __attribute__((ext_vector_type(8)))  float  v8f;

union Frag16 {
    v16bf v;
    uint4 q[2];
    unsigned short h[16];
};

static __device__ __forceinline__ v8f wmma_bf16(const v16bf& a, const v16bf& b, const v8f& c) {
    return __builtin_amdgcn_wmma_f32_16x16x32_bf16(false, a, false, b, (short)0, c, false, false);
}

static __device__ __forceinline__ unsigned short f2bfu(float f) {
    unsigned u = __builtin_bit_cast(unsigned, f);
    u = (u + 0x7FFFu + ((u >> 16) & 1u)) >> 16;
    return (unsigned short)u;
}

// Problem constants
#define BATCH 4
#define NTOK  2048
#define PD    512
#define CD    128
#define ID    512
#define NH    8
#define DH    64

// ---------------------------------------------------------------------------
// Fragment loaders (all contiguous 16B/32B global or LDS loads).
// A-frag (16x32, bf16): lane l<16 -> row l, K {k0..k0+7, k0+16..k0+23}
//                       lane l>=16 -> row l-16, K {k0+8..15, k0+24..31}
// B-frag (32x16, bf16): lane l<16 -> col l, K {k0..k0+7, k0+16..k0+23}
//                       lane l>=16 -> col l-16, K {k0+8..15, k0+24..31}
// ---------------------------------------------------------------------------
static __device__ __forceinline__ void load_afrag_u16(Frag16& a, const unsigned short* base, int ld,
                                                      int row0, int k0, int l, int g) {
    const unsigned short* p = base + (size_t)(row0 + l) * ld + k0 + 8 * g;
    a.q[0] = *(const uint4*)(p);
    a.q[1] = *(const uint4*)(p + 16);
}

static __device__ __forceinline__ void load_bfrag_u16(Frag16& b, const unsigned short* base, int ld,
                                                      int n0, int k0, int l, int g) {
    const unsigned short* p = base + (size_t)(n0 + l) * ld + k0 + 8 * g;
    b.q[0] = *(const uint4*)(p);
    b.q[1] = *(const uint4*)(p + 16);
}

// ---------------------------------------------------------------------------
// Kernel: elementwise fp32 -> bf16 (vectorized, bandwidth-bound, one-off)
// ---------------------------------------------------------------------------
__global__ void convert_bf16_kernel(const float* __restrict__ src, unsigned short* __restrict__ dst,
                                    int n4) {
    int i = blockIdx.x * blockDim.x + threadIdx.x;
    if (i >= n4) return;
    float4 x = ((const float4*)src)[i];
    ushort4 o;
    o.x = f2bfu(x.x); o.y = f2bfu(x.y); o.z = f2bfu(x.z); o.w = f2bfu(x.w);
    ((ushort4*)dst)[i] = o;
}

// ---------------------------------------------------------------------------
// Kernel: transpose fp32 weight [K][NC] -> bf16 [NC][K] (one-off, tiny)
// ---------------------------------------------------------------------------
__global__ void transpose_w_kernel(const float* __restrict__ W, unsigned short* __restrict__ Wt,
                                   int K, int NC) {
    int idx = blockIdx.x * blockDim.x + threadIdx.x;
    if (idx >= K * NC) return;
    int n = idx / K;
    int k = idx - n * K;
    Wt[idx] = f2bfu(W[(size_t)k * NC + n]);
}

// ---------------------------------------------------------------------------
// QKV projection epilogue for one 16x16 tile: split into qp / kp (clamp +-5)
// per-head layout [b][h][n][64], and v transposed [b][h][64][n].
// ---------------------------------------------------------------------------
static __device__ __forceinline__ void store_qkv_tile(const v8f& c, int row0, int col0, int l, int g,
                                                      unsigned short* __restrict__ qp,
                                                      unsigned short* __restrict__ kp,
                                                      unsigned short* __restrict__ vT) {
#pragma unroll
    for (int r = 0; r < 8; r++) {
        int row = row0 + r + 8 * g;
        int col = col0 + l;
        int bb = row >> 11, tok = row & (NTOK - 1);
        float v = c[r];
        if (col < ID) {
            float cv = fminf(fmaxf(v, -5.f), 5.f);
            int hh = col >> 6, d = col & 63;
            qp[((size_t)(bb * NH + hh) * NTOK + tok) * DH + d] = f2bfu(cv);
        } else if (col < 2 * ID) {
            int cc = col - ID;
            float cv = fminf(fmaxf(v, -5.f), 5.f);
            int hh = cc >> 6, d = cc & 63;
            kp[((size_t)(bb * NH + hh) * NTOK + tok) * DH + d] = f2bfu(cv);
        } else {
            int cc = col - 2 * ID;
            int hh = cc >> 6, d = cc & 63;
            vT[((size_t)(bb * NH + hh) * DH + d) * NTOK + tok] = f2bfu(v);
        }
    }
}

// ---------------------------------------------------------------------------
// Kernel: pixels_bf16(8192x512) @ Wqkv_t -> qp, kp, vT.
// One 32x32 output block per wave (2x2 WMMA tiles: 4 WMMAs per 4 frag loads).
// K-loop unroll capped at 2 to avoid register spills (8 frags + 4 accum live).
// ---------------------------------------------------------------------------
__global__ __launch_bounds__(256) void proj_qkv_kernel(const unsigned short* __restrict__ A,
                                                       const unsigned short* __restrict__ Wt,
                                                       unsigned short* __restrict__ qp,
                                                       unsigned short* __restrict__ kp,
                                                       unsigned short* __restrict__ vT) {
    const int NT = (3 * ID) / 32;  // 48 column blocks
    int w = blockIdx.x * 8 + (threadIdx.x >> 5);
    int lane = threadIdx.x & 31, l = lane & 15, g = lane >> 4;
    int mt = w / NT, nt = w - mt * NT;
    int m0 = mt * 32, n0 = nt * 32;

    v8f c00 = {}, c01 = {}, c10 = {}, c11 = {};
#pragma unroll 2
    for (int k0 = 0; k0 < PD; k0 += 32) {
        Frag16 a0, a1, b0, b1;
        load_afrag_u16(a0, A, PD, m0, k0, l, g);
        load_afrag_u16(a1, A, PD, m0 + 16, k0, l, g);
        load_bfrag_u16(b0, Wt, PD, n0, k0, l, g);
        load_bfrag_u16(b1, Wt, PD, n0 + 16, k0, l, g);
        c00 = wmma_bf16(a0.v, b0.v, c00);
        c01 = wmma_bf16(a0.v, b1.v, c01);
        c10 = wmma_bf16(a1.v, b0.v, c10);
        c11 = wmma_bf16(a1.v, b1.v, c11);
    }
    store_qkv_tile(c00, m0,      n0,      l, g, qp, kp, vT);
    store_qkv_tile(c01, m0,      n0 + 16, l, g, qp, kp, vT);
    store_qkv_tile(c10, m0 + 16, n0,      l, g, qp, kp, vT);
    store_qkv_tile(c11, m0 + 16, n0 + 16, l, g, qp, kp, vT);
}

// ---------------------------------------------------------------------------
// QcKc projection epilogue for one 16x16 tile
// ---------------------------------------------------------------------------
static __device__ __forceinline__ void store_qkc_tile(const v8f& c, int row0, int col0, int l, int g,
                                                      unsigned short* __restrict__ qc,
                                                      unsigned short* __restrict__ kc) {
#pragma unroll
    for (int r = 0; r < 8; r++) {
        int row = row0 + r + 8 * g;
        int col = col0 + l;
        int bb = row >> 11, tok = row & (NTOK - 1);
        float cv = fminf(fmaxf(c[r], -5.f), 5.f);
        int cc = (col < ID) ? col : (col - ID);
        int hh = cc >> 6, d = cc & 63;
        unsigned short* dst = (col < ID) ? qc : kc;
        dst[((size_t)(bb * NH + hh) * NTOK + tok) * DH + d] = f2bfu(cv);
    }
}

// ---------------------------------------------------------------------------
// Kernel: coords_bf16(8192x128) @ Wqkc_t -> qc, kc. 32x32 block per wave.
// ---------------------------------------------------------------------------
__global__ __launch_bounds__(256) void proj_qkc_kernel(const unsigned short* __restrict__ A,
                                                       const unsigned short* __restrict__ Wt,
                                                       unsigned short* __restrict__ qc,
                                                       unsigned short* __restrict__ kc) {
    const int NT = (2 * ID) / 32;  // 32 column blocks
    int w = blockIdx.x * 8 + (threadIdx.x >> 5);
    int lane = threadIdx.x & 31, l = lane & 15, g = lane >> 4;
    int mt = w / NT, nt = w - mt * NT;
    int m0 = mt * 32, n0 = nt * 32;

    v8f c00 = {}, c01 = {}, c10 = {}, c11 = {};
#pragma unroll 2
    for (int k0 = 0; k0 < CD; k0 += 32) {
        Frag16 a0, a1, b0, b1;
        load_afrag_u16(a0, A, CD, m0, k0, l, g);
        load_afrag_u16(a1, A, CD, m0 + 16, k0, l, g);
        load_bfrag_u16(b0, Wt, CD, n0, k0, l, g);
        load_bfrag_u16(b1, Wt, CD, n0 + 16, k0, l, g);
        c00 = wmma_bf16(a0.v, b0.v, c00);
        c01 = wmma_bf16(a0.v, b1.v, c01);
        c10 = wmma_bf16(a1.v, b0.v, c10);
        c11 = wmma_bf16(a1.v, b1.v, c11);
    }
    store_qkc_tile(c00, m0,      n0,      l, g, qc, kc);
    store_qkc_tile(c01, m0,      n0 + 16, l, g, qc, kc);
    store_qkc_tile(c10, m0 + 16, n0,      l, g, qc, kc);
    store_qkc_tile(c11, m0 + 16, n0 + 16, l, g, qc, kc);
}

// ---------------------------------------------------------------------------
// Kernel: flash attention. One wave handles 16 query rows of one (b,h);
// streams keys in chunks of 32 with online softmax. 12 WMMAs per chunk.
// Writes O (bf16) as [b][n][h*64] for the output projection.
// ---------------------------------------------------------------------------
__global__ __launch_bounds__(256) void attn_kernel(const unsigned short* __restrict__ qp,
                                                   const unsigned short* __restrict__ kp,
                                                   const unsigned short* __restrict__ qc,
                                                   const unsigned short* __restrict__ kc,
                                                   const unsigned short* __restrict__ vT,
                                                   const unsigned char* __restrict__ mask,
                                                   unsigned short* __restrict__ Ob) {
    __shared__ unsigned short pbuf[8][16 * 32];  // per-wave P staging (C-layout -> A-layout)

    int wave = threadIdx.x >> 5;
    int lane = threadIdx.x & 31, l = lane & 15, g = lane >> 4;
    int bh = blockIdx.x >> 4;       // 32 (b,h) pairs
    int qblk = blockIdx.x & 15;     // 16 query blocks of 128
    int b = bh >> 3, h = bh & 7;
    int q0 = qblk * 128 + wave * 16;

    const unsigned short* qpb = qp + (size_t)bh * NTOK * DH;
    const unsigned short* kpb = kp + (size_t)bh * NTOK * DH;
    const unsigned short* qcb = qc + (size_t)bh * NTOK * DH;
    const unsigned short* kcb = kc + (size_t)bh * NTOK * DH;
    const unsigned short* vTb = vT + (size_t)bh * DH * NTOK;
    const unsigned char*  mb  = mask + (size_t)b * NTOK;

    // Q fragments (held in registers for the whole K loop)
    Frag16 aqp0, aqp1, aqc0, aqc1;
    load_afrag_u16(aqp0, qpb, DH, q0, 0, l, g);
    load_afrag_u16(aqp1, qpb, DH, q0, 32, l, g);
    load_afrag_u16(aqc0, qcb, DH, q0, 0, l, g);
    load_afrag_u16(aqc1, qcb, DH, q0, 32, l, g);

    v8f o[4];
    o[0] = v8f{}; o[1] = v8f{}; o[2] = v8f{}; o[3] = v8f{};
    float mrow[8], lrow[8];
#pragma unroll
    for (int r = 0; r < 8; r++) { mrow[r] = -1e30f; lrow[r] = 0.f; }

    const float scale = 0.125f;  // 64^-0.5

#pragma unroll 1
    for (int j = 0; j < NTOK; j += 32) {
        Frag16 f0, f1;
        v8f s0 = {}, s1 = {};
        // S tile 0: keys j..j+15
        load_bfrag_u16(f0, kpb, DH, j, 0, l, g);
        load_bfrag_u16(f1, kpb, DH, j, 32, l, g);
        s0 = wmma_bf16(aqp0.v, f0.v, s0);
        s0 = wmma_bf16(aqp1.v, f1.v, s0);
        load_bfrag_u16(f0, kcb, DH, j, 0, l, g);
        load_bfrag_u16(f1, kcb, DH, j, 32, l, g);
        s0 = wmma_bf16(aqc0.v, f0.v, s0);
        s0 = wmma_bf16(aqc1.v, f1.v, s0);
        // S tile 1: keys j+16..j+31
        load_bfrag_u16(f0, kpb, DH, j + 16, 0, l, g);
        load_bfrag_u16(f1, kpb, DH, j + 16, 32, l, g);
        s1 = wmma_bf16(aqp0.v, f0.v, s1);
        s1 = wmma_bf16(aqp1.v, f1.v, s1);
        load_bfrag_u16(f0, kcb, DH, j + 16, 0, l, g);
        load_bfrag_u16(f1, kcb, DH, j + 16, 32, l, g);
        s1 = wmma_bf16(aqc0.v, f0.v, s1);
        s1 = wmma_bf16(aqc1.v, f1.v, s1);

        float msk0 = mb[j + l]      ? -3e38f : 0.f;
        float msk1 = mb[j + 16 + l] ? -3e38f : 0.f;

        // online softmax update (rows r and r+8 live in lane halves)
#pragma unroll
        for (int r = 0; r < 8; r++) {
            float a0 = s0[r] * scale + msk0;
            float a1 = s1[r] * scale + msk1;
            float tmax = fmaxf(a0, a1);
            tmax = fmaxf(tmax, __shfl_xor(tmax, 1, 32));
            tmax = fmaxf(tmax, __shfl_xor(tmax, 2, 32));
            tmax = fmaxf(tmax, __shfl_xor(tmax, 4, 32));
            tmax = fmaxf(tmax, __shfl_xor(tmax, 8, 32));
            float mn = fmaxf(mrow[r], tmax);
            float alpha = __expf(mrow[r] - mn);
            mrow[r] = mn;
            float p0 = __expf(a0 - mn);
            float p1 = __expf(a1 - mn);
            float rs = p0 + p1;
            rs += __shfl_xor(rs, 1, 32);
            rs += __shfl_xor(rs, 2, 32);
            rs += __shfl_xor(rs, 4, 32);
            rs += __shfl_xor(rs, 8, 32);
            lrow[r] = lrow[r] * alpha + rs;
            o[0][r] *= alpha; o[1][r] *= alpha; o[2][r] *= alpha; o[3][r] *= alpha;
            pbuf[wave][(r + 8 * g) * 32 + l]      = f2bfu(p0);
            pbuf[wave][(r + 8 * g) * 32 + 16 + l] = f2bfu(p1);
        }

        // P as A-fragment (16x32 over this key chunk)
        Frag16 pa;
        pa.q[0] = *(const uint4*)&pbuf[wave][l * 32 + 8 * g];
        pa.q[1] = *(const uint4*)&pbuf[wave][l * 32 + 8 * g + 16];

        // O += P @ V  (V transposed layout gives contiguous B-fragments)
#pragma unroll
        for (int t = 0; t < 4; t++) {
            Frag16 vf;
            const unsigned short* p = vTb + (size_t)(16 * t + l) * NTOK + j + 8 * g;
            vf.q[0] = *(const uint4*)(p);
            vf.q[1] = *(const uint4*)(p + 16);
            o[t] = wmma_bf16(pa.v, vf.v, o[t]);
        }
    }

    // finalize: divide by row sums, write to [b][n][h*64] (bf16)
#pragma unroll
    for (int r = 0; r < 8; r++) {
        float inv = 1.f / lrow[r];
        int row = q0 + r + 8 * g;
        size_t base = ((size_t)(b * NTOK + row)) * ID + h * DH + l;
        Ob[base + 0]  = f2bfu(o[0][r] * inv);
        Ob[base + 16] = f2bfu(o[1][r] * inv);
        Ob[base + 32] = f2bfu(o[2][r] * inv);
        Ob[base + 48] = f2bfu(o[3][r] * inv);
    }
}

// ---------------------------------------------------------------------------
// Kernel: O(8192x512, bf16) @ Wout_t + b_out -> fp32 out, with two-way
// sequence split. 32x32 block per wave.
// ---------------------------------------------------------------------------
__global__ __launch_bounds__(256) void outproj_kernel(const unsigned short* __restrict__ Ob,
                                                      const unsigned short* __restrict__ Wt,
                                                      const float* __restrict__ bout,
                                                      float* __restrict__ out) {
    const int NT = PD / 32;  // 16 column blocks
    int w = blockIdx.x * 8 + (threadIdx.x >> 5);
    int lane = threadIdx.x & 31, l = lane & 15, g = lane >> 4;
    int mt = w / NT, nt = w - mt * NT;
    int m0 = mt * 32, n0 = nt * 32;

    v8f c[2][2];
    c[0][0] = v8f{}; c[0][1] = v8f{}; c[1][0] = v8f{}; c[1][1] = v8f{};
#pragma unroll 2
    for (int k0 = 0; k0 < ID; k0 += 32) {
        Frag16 a0, a1, b0, b1;
        load_afrag_u16(a0, Ob, ID, m0, k0, l, g);
        load_afrag_u16(a1, Ob, ID, m0 + 16, k0, l, g);
        load_bfrag_u16(b0, Wt, ID, n0, k0, l, g);
        load_bfrag_u16(b1, Wt, ID, n0 + 16, k0, l, g);
        c[0][0] = wmma_bf16(a0.v, b0.v, c[0][0]);
        c[0][1] = wmma_bf16(a0.v, b1.v, c[0][1]);
        c[1][0] = wmma_bf16(a1.v, b0.v, c[1][0]);
        c[1][1] = wmma_bf16(a1.v, b1.v, c[1][1]);
    }
#pragma unroll
    for (int ti = 0; ti < 2; ti++) {
#pragma unroll
        for (int tj = 0; tj < 2; tj++) {
#pragma unroll
            for (int r = 0; r < 8; r++) {
                int row = m0 + 16 * ti + r + 8 * g;
                int col = n0 + 16 * tj + l;
                int bb = row >> 11, n = row & (NTOK - 1);
                float v = c[ti][tj][r] + bout[col];
                size_t idx;
                if (n < 1024) {
                    idx = ((size_t)(bb * 1024 + n)) * PD + col;
                } else {
                    idx = (size_t)BATCH * 1024 * PD + ((size_t)(bb * 1024 + (n - 1024))) * PD + col;
                }
                out[idx] = v;
            }
        }
    }
}

// ---------------------------------------------------------------------------
extern "C" void kernel_launch(void* const* d_in, const int* in_sizes, int n_in,
                              void* d_out, int out_size, void* d_ws, size_t ws_size,
                              hipStream_t stream) {
    const float* pixels = (const float*)d_in[0];          // (4,2048,512)
    const float* coords = (const float*)d_in[1];          // (4,2048,128)
    const unsigned char* mask = (const unsigned char*)d_in[2];  // (4,2048) bool
    const float* W_qkv = (const float*)d_in[3];           // (512,1536)
    const float* W_qkc = (const float*)d_in[4];           // (128,1024)
    const float* W_out = (const float*)d_in[5];           // (512,512)
    const float* b_out = (const float*)d_in[6];           // (512,)
    float* out = (float*)d_out;

    // workspace layout (bf16 element counts)
    unsigned short* ws = (unsigned short*)d_ws;
    const size_t HEADS_ELEMS = (size_t)BATCH * NH * NTOK * DH;  // 4M
    unsigned short* qp    = ws;
    unsigned short* kp    = qp + HEADS_ELEMS;
    unsigned short* qc    = kp + HEADS_ELEMS;
    unsigned short* kc    = qc + HEADS_ELEMS;
    unsigned short* vT    = kc + HEADS_ELEMS;
    unsigned short* Obuf  = vT + HEADS_ELEMS;
    unsigned short* pixB  = Obuf + (size_t)BATCH * NTOK * ID;
    unsigned short* cooB  = pixB + (size_t)BATCH * NTOK * PD;
    unsigned short* Wqkvt = cooB + (size_t)BATCH * NTOK * CD;
    unsigned short* Wqkct = Wqkvt + (size_t)PD * 3 * ID;
    unsigned short* Woutt = Wqkct + (size_t)CD * 2 * ID;

    // 1) one-off conversions: activations fp32->bf16, weights transpose+convert
    {
        int n4 = (BATCH * NTOK * PD) / 4;
        convert_bf16_kernel<<<(n4 + 255) / 256, 256, 0, stream>>>(pixels, pixB, n4);
    }
    {
        int n4 = (BATCH * NTOK * CD) / 4;
        convert_bf16_kernel<<<(n4 + 255) / 256, 256, 0, stream>>>(coords, cooB, n4);
    }
    transpose_w_kernel<<<(PD * 3 * ID + 255) / 256, 256, 0, stream>>>(W_qkv, Wqkvt, PD, 3 * ID);
    transpose_w_kernel<<<(CD * 2 * ID + 255) / 256, 256, 0, stream>>>(W_qkc, Wqkct, CD, 2 * ID);
    transpose_w_kernel<<<(ID * PD + 255) / 256, 256, 0, stream>>>(W_out, Woutt, ID, PD);

    // 2) projections (32x32 block per wave)
    {
        int blocks = ((BATCH * NTOK / 32) * ((3 * ID) / 32)) / 8;  // 1536
        proj_qkv_kernel<<<blocks, 256, 0, stream>>>(pixB, Wqkvt, qp, kp, vT);
    }
    {
        int blocks = ((BATCH * NTOK / 32) * ((2 * ID) / 32)) / 8;  // 1024
        proj_qkc_kernel<<<blocks, 256, 0, stream>>>(cooB, Wqkct, qc, kc);
    }

    // 3) flash attention: B*H (32) x 16 query blocks of 128 rows
    attn_kernel<<<BATCH * NH * 16, 256, 0, stream>>>(qp, kp, qc, kc, vT, mask, Obuf);

    // 4) output projection + bias + sequence split
    {
        int blocks = ((BATCH * NTOK / 32) * (PD / 32)) / 8;  // 512
        outproj_kernel<<<blocks, 256, 0, stream>>>(Obuf, Woutt, b_out, out);
    }
}
